// RiverNetworkGNN_21680994910369
// MI455X (gfx1250) — matrix operand order repaired and checked by reading, not compiled, not run on hardware
//
#include <hip/hip_runtime.h>
#include <hip/hip_bf16.h>
#include <math.h>

// ---------------------------------------------------------------------------
// RiverNetworkGNN for MI455X (gfx1250): bf16 WMMA GEMMs + fused message pass
// v2: 2 M-tiles per wave sharing B fragments (12 loads : 8 wmma per k-step)
// ---------------------------------------------------------------------------

typedef __attribute__((ext_vector_type(16))) __bf16 v16bf;
typedef __attribute__((ext_vector_type(8)))  __bf16 v8bf;
typedef __attribute__((ext_vector_type(8)))  float  v8f;

union ABf { v16bf full; v8bf half[2]; };

#define N_NODES 50000
#define N_EDGES 150000
#define HID     512

__device__ __forceinline__ __bf16 f2bf(float f) {
  unsigned u = __builtin_bit_cast(unsigned, f);
  u += 0x7FFFu + ((u >> 16) & 1u);            // round to nearest even
  unsigned short h = (unsigned short)(u >> 16);
  return __builtin_bit_cast(__bf16, h);
}

__device__ __forceinline__ float gelu_exact(float x) {
  return 0.5f * x * (1.0f + erff(x * 0.70710678118654752f));
}

// ---- prep kernels ---------------------------------------------------------

// dst[n*K + k] = bf16(src[k*N + n])   (weight transpose + downconvert)
__global__ void k_transpose_bf16(const float* __restrict__ src,
                                 __bf16* __restrict__ dst, int K, int N) {
  int idx = blockIdx.x * blockDim.x + threadIdx.x;
  if (idx >= K * N) return;
  int k = idx / N;
  int n = idx - k * N;
  dst[(size_t)n * K + k] = f2bf(src[idx]);
}

__global__ void k_f32_to_bf16(const float* __restrict__ src,
                              __bf16* __restrict__ dst, int n) {
  int idx = blockIdx.x * blockDim.x + threadIdx.x;
  if (idx < n) dst[idx] = f2bf(src[idx]);
}

// ---- dense GEMM: out = act(A[MxK] @ W + bias (+ river@Wr)) ----------------
// 256 threads = 8 waves; block -> 32 rows x 512 cols; wave w -> cols [64w,64w+64)
// Each wave: 2 M-tiles x 4 N-tiles, B fragments shared across M-tiles.
// MODE 0: layer1 (add rank-3 river term, GELU, bf16 out)
// MODE 1: layer2 (f32 out)
template <int MODE>
__global__ void k_gemm(const __bf16* __restrict__ A,     // [M x 512] bf16 row-major
                       const __bf16* __restrict__ Bt,    // [512 x 512] = W^T (N x K)
                       const float*  __restrict__ bias,  // [512]
                       const float*  __restrict__ river, // [M x 3] (MODE 0)
                       const float*  __restrict__ Wr,    // rows 512..514 of W1 (MODE 0)
                       void* __restrict__ out, int M)
{
  const int K    = HID;
  const int row0 = blockIdx.x * 32;
  const int lane = threadIdx.x & 31;
  const int wave = threadIdx.x >> 5;
  const int r    = lane & 15;
  const int hi   = lane >> 4;          // 0 or 1
  const int kb   = hi * 8;             // A-fragment K base
  const int kb2  = hi * 16;            // unused; B fragment also uses kb
  const int nb   = wave * 64;

  int ra = row0 + r;
  int rb = row0 + 16 + r;
  const int rac = (ra < M) ? ra : (M - 1);   // clamp for tail block
  const int rbc = (rb < M) ? rb : (M - 1);
  const __bf16* arow0 = A + (size_t)rac * K;
  const __bf16* arow1 = A + (size_t)rbc * K;

  v8f acc[2][4] = {};

  for (int k0 = 0; k0 < K; k0 += 32) {
    ABf a0, a1;
    a0.half[0] = *(const v8bf*)(arow0 + k0 + kb);
    a0.half[1] = *(const v8bf*)(arow0 + k0 + 16 + kb);
    a1.half[0] = *(const v8bf*)(arow1 + k0 + kb);
    a1.half[1] = *(const v8bf*)(arow1 + k0 + 16 + kb);
#pragma unroll
    for (int t = 0; t < 4; ++t) {
      const __bf16* brow = Bt + (size_t)(nb + 16 * t + r) * K + k0 + kb;
      ABf b;
      b.half[0] = *(const v8bf*)(brow);
      b.half[1] = *(const v8bf*)(brow + 16);
      acc[0][t] = __builtin_amdgcn_wmma_f32_16x16x32_bf16(
          false, a0.full, false, b.full, (short)0, acc[0][t], false, false);
      acc[1][t] = __builtin_amdgcn_wmma_f32_16x16x32_bf16(
          false, a1.full, false, b.full, (short)0, acc[1][t], false, false);
    }
  }

#pragma unroll
  for (int mt = 0; mt < 2; ++mt) {
#pragma unroll
    for (int t = 0; t < 4; ++t) {
      const int n  = nb + 16 * t + r;
      const float bn = bias[n];
      float w0 = 0.f, w1 = 0.f, w2 = 0.f;
      if (MODE == 0) { w0 = Wr[n]; w1 = Wr[HID + n]; w2 = Wr[2 * HID + n]; }
#pragma unroll
      for (int j = 0; j < 8; ++j) {
        const int m = row0 + mt * 16 + j + hi * 8;
        if (m < M) {
          float v = acc[mt][t][j] + bn;
          if (MODE == 0) {
            const float* rv = river + (size_t)m * 3;
            v += rv[0] * w0 + rv[1] * w1 + rv[2] * w2;
            v = gelu_exact(v);
            ((__bf16*)out)[(size_t)m * HID + n] = f2bf(v);
          } else {
            ((float*)out)[(size_t)m * HID + n] = v;
          }
        }
      }
    }
  }
}

// ---- fused message pass ---------------------------------------------------
// block = 32 edges, 256 threads (8 waves, 64 cols each, 2 M-tiles per wave)
// h = gelu([F[src]||F[tgt]] @ Wm1 + bm1)  -> LDS (bf16)
// msg = h @ Wm2 + bm2 ; F[tgt] += msg  (segment_sum via f32 atomics)
__global__ void k_msgpass(const __bf16*   __restrict__ Fb,    // [N_NODES x 512] bf16 snapshot
                          const long long* __restrict__ edges, // [E x 2] int64
                          const __bf16*   __restrict__ Wm1t,  // [512 x 1024]
                          const float*    __restrict__ bm1,
                          const __bf16*   __restrict__ Wm2t,  // [512 x 512]
                          const float*    __restrict__ bm2,
                          float* __restrict__ F)               // in-place accumulate
{
  __shared__ __attribute__((aligned(16))) __bf16 h_lds[32 * HID]; // 32 KB
  __shared__ int tgt_l[32];

  const int e0   = blockIdx.x * 32;
  const int lane = threadIdx.x & 31;
  const int wave = threadIdx.x >> 5;
  const int r    = lane & 15;
  const int hi   = lane >> 4;
  const int kb   = hi * 8;
  const int kb2  = hi * 16;
  const int nb   = wave * 64;

  // per-lane edge rows for the two M-tiles (clamped for the tail block)
  const int ea = e0 + r;
  const int eb = e0 + 16 + r;
  const size_t eac = (size_t)((ea < N_EDGES) ? ea : (N_EDGES - 1));
  const size_t ebc = (size_t)((eb < N_EDGES) ? eb : (N_EDGES - 1));
  const int sid0 = (int)edges[2 * eac];
  const int tid0 = (int)edges[2 * eac + 1];
  const int sid1 = (int)edges[2 * ebc];
  const int tid1 = (int)edges[2 * ebc + 1];
  if (threadIdx.x < 32) {
    const int e = e0 + (int)threadIdx.x;
    const size_t ec = (size_t)((e < N_EDGES) ? e : (N_EDGES - 1));
    tgt_l[threadIdx.x] = (int)edges[2 * ec + 1];
  }

  const __bf16* srow0 = Fb + (size_t)sid0 * HID;
  const __bf16* trow0 = Fb + (size_t)tid0 * HID;
  const __bf16* srow1 = Fb + (size_t)sid1 * HID;
  const __bf16* trow1 = Fb + (size_t)tid1 * HID;

  // stage 1: h = gelu(concat @ Wm1 + bm1), K = 1024
  {
    v8f acc[2][4] = {};
    for (int k0 = 0; k0 < 2 * HID; k0 += 32) {
      const __bf16* arow0 = (k0 < HID) ? (srow0 + k0) : (trow0 + (k0 - HID));
      const __bf16* arow1 = (k0 < HID) ? (srow1 + k0) : (trow1 + (k0 - HID));
      ABf a0, a1;
      a0.half[0] = *(const v8bf*)(arow0 + kb);
      a0.half[1] = *(const v8bf*)(arow0 + 16 + kb);
      a1.half[0] = *(const v8bf*)(arow1 + kb);
      a1.half[1] = *(const v8bf*)(arow1 + 16 + kb);
#pragma unroll
      for (int t = 0; t < 4; ++t) {
        const __bf16* brow = Wm1t + (size_t)(nb + 16 * t + r) * (2 * HID) + k0 + kb;
        ABf b;
        b.half[0] = *(const v8bf*)(brow);
        b.half[1] = *(const v8bf*)(brow + 16);
        acc[0][t] = __builtin_amdgcn_wmma_f32_16x16x32_bf16(
            false, a0.full, false, b.full, (short)0, acc[0][t], false, false);
        acc[1][t] = __builtin_amdgcn_wmma_f32_16x16x32_bf16(
            false, a1.full, false, b.full, (short)0, acc[1][t], false, false);
      }
    }
#pragma unroll
    for (int mt = 0; mt < 2; ++mt) {
#pragma unroll
      for (int t = 0; t < 4; ++t) {
        const int n  = nb + 16 * t + r;
        const float bn = bm1[n];
#pragma unroll
        for (int j = 0; j < 8; ++j) {
          const int m = mt * 16 + j + hi * 8;
          h_lds[m * HID + n] = f2bf(gelu_exact(acc[mt][t][j] + bn));
        }
      }
    }
  }
  __syncthreads();

  // stage 2: msg = h @ Wm2 + bm2, K = 512; scatter-add into F[tgt]
  v8f acc2[2][4] = {};
  for (int k0 = 0; k0 < HID; k0 += 32) {
    const __bf16* arow0 = &h_lds[r * HID + k0];
    const __bf16* arow1 = &h_lds[(16 + r) * HID + k0];
    ABf a0, a1;
    a0.half[0] = *(const v8bf*)(arow0 + kb);
    a0.half[1] = *(const v8bf*)(arow0 + 16 + kb);
    a1.half[0] = *(const v8bf*)(arow1 + kb);
    a1.half[1] = *(const v8bf*)(arow1 + 16 + kb);
#pragma unroll
    for (int t = 0; t < 4; ++t) {
      const __bf16* brow = Wm2t + (size_t)(nb + 16 * t + r) * HID + k0 + kb;
      ABf b;
      b.half[0] = *(const v8bf*)(brow);
      b.half[1] = *(const v8bf*)(brow + 16);
      acc2[0][t] = __builtin_amdgcn_wmma_f32_16x16x32_bf16(
          false, a0.full, false, b.full, (short)0, acc2[0][t], false, false);
      acc2[1][t] = __builtin_amdgcn_wmma_f32_16x16x32_bf16(
          false, a1.full, false, b.full, (short)0, acc2[1][t], false, false);
    }
  }
#pragma unroll
  for (int mt = 0; mt < 2; ++mt) {
#pragma unroll
    for (int t = 0; t < 4; ++t) {
      const int n  = nb + 16 * t + r;
      const float bn = bm2[n];
#pragma unroll
      for (int j = 0; j < 8; ++j) {
        const int m = mt * 16 + j + hi * 8;
        if (e0 + m < N_EDGES) {
          const int tg = tgt_l[m];
          atomicAdd(&F[(size_t)tg * HID + n], acc2[mt][t][j] + bn);
        }
      }
    }
  }
}

// ---------------------------------------------------------------------------

extern "C" void kernel_launch(void* const* d_in, const int* in_sizes, int n_in,
                              void* d_out, int out_size, void* d_ws, size_t ws_size,
                              hipStream_t stream) {
  (void)in_sizes; (void)n_in; (void)out_size; (void)ws_size;

  const float*     node  = (const float*)d_in[0];
  const float*     river = (const float*)d_in[1];
  const long long* edges = (const long long*)d_in[2];
  const float*     W1    = (const float*)d_in[3];
  const float*     b1    = (const float*)d_in[4];
  const float*     W2    = (const float*)d_in[5];
  const float*     b2    = (const float*)d_in[6];
  const float*     Wm1   = (const float*)d_in[7];
  const float*     bm1   = (const float*)d_in[8];
  const float*     Wm2   = (const float*)d_in[9];
  const float*     bm2   = (const float*)d_in[10];

  char* ws = (char*)d_ws;
  __bf16* W1t  = (__bf16*)(ws);                             // 512x512  (0.5 MB)
  __bf16* W2t  = (__bf16*)(ws + (512u * 512u * 2u));        // 512x512
  __bf16* Wm1t = (__bf16*)(ws + (2u * 512u * 512u * 2u));   // 512x1024 (1 MB)
  __bf16* Wm2t = (__bf16*)(ws + (4u * 512u * 512u * 2u));   // 512x512
  size_t  woff = 5u * 512u * 512u * 2u;                     // 2.5 MB
  __bf16* Xb   = (__bf16*)(ws + woff);                      // 50000x512 bf16 (reused as Fb)
  __bf16* Hb   = (__bf16*)(ws + woff + (size_t)N_NODES * HID * 2u);
  float*  F    = (float*)d_out;

  const int NELT = N_NODES * HID;  // 25,600,000
  const int GEMM_BLOCKS = (N_NODES + 31) / 32;   // 1563
  const int MSG_BLOCKS  = (N_EDGES + 31) / 32;   // 4688

  // weight prep (bf16, transposed to [N x K])
  k_transpose_bf16<<<(512 * 512 + 255) / 256, 256, 0, stream>>>(W1, W1t, 512, 512);
  k_transpose_bf16<<<(512 * 512 + 255) / 256, 256, 0, stream>>>(W2, W2t, 512, 512);
  k_transpose_bf16<<<(1024 * 512 + 255) / 256, 256, 0, stream>>>(Wm1, Wm1t, 1024, 512);
  k_transpose_bf16<<<(512 * 512 + 255) / 256, 256, 0, stream>>>(Wm2, Wm2t, 512, 512);
  k_f32_to_bf16<<<(NELT + 255) / 256, 256, 0, stream>>>(node, Xb, NELT);

  // encoder: H = gelu(X@W1 + river@W1r + b1);  F = H@W2 + b2
  k_gemm<0><<<GEMM_BLOCKS, 256, 0, stream>>>(Xb, W1t, b1, river, W1 + 512 * 512, Hb, N_NODES);
  k_gemm<1><<<GEMM_BLOCKS, 256, 0, stream>>>(Hb, W2t, b2, nullptr, nullptr, F, N_NODES);

  // 3 message-passing rounds (snapshot -> fused gather/GEMM/GEMM/scatter-add)
  for (int p = 0; p < 3; ++p) {
    k_f32_to_bf16<<<(NELT + 255) / 256, 256, 0, stream>>>(F, Xb, NELT);
    k_msgpass<<<MSG_BLOCKS, 256, 0, stream>>>(Xb, edges, Wm1t, bm1, Wm2t, bm2, F);
  }
}
